// ApproxROT_85718957294045
// MI455X (gfx1250) — matrix operand order repaired and checked by reading, not compiled, not run on hardware
//
#include <hip/hip_runtime.h>

typedef _Float16 v16h __attribute__((ext_vector_type(16)));
typedef _Float16 v8h  __attribute__((ext_vector_type(8)));
typedef float    v8f  __attribute__((ext_vector_type(8)));
typedef unsigned int v4u __attribute__((ext_vector_type(4)));
typedef int v8i __attribute__((ext_vector_type(8)));
typedef int v4i __attribute__((ext_vector_type(4)));

#define B_  8
#define N_  1024
#define D_  512
#define EPSV 1e-8f
#define BND (B_*N_*D_)

#if defined(__gfx1250__) && __has_builtin(__builtin_amdgcn_tensor_load_to_lds) && __has_builtin(__builtin_amdgcn_s_wait_tensorcnt)
#define USE_TDM 1
#endif

#define LDS_STRIDE 40          // 32 halfs data + 8 halfs pad (64B + 16B)
#define TILE_HALFS (128 * LDS_STRIDE)
#define TILE_BYTES (TILE_HALFS * 2)

#ifdef USE_TDM
// LDS offset (bytes) of a __shared__ pointer: generic -> addrspace(3) -> int
__device__ __forceinline__ unsigned lds_off(const void* p) {
  return (unsigned)(unsigned long long)(__attribute__((address_space(3))) const char*)p;
}

// Issue one TDM 2-D tile load: 128 rows x 32 halfs, row stride = rowStride
// elements, LDS rows padded to 40 halfs via pad_interval=3 (16 DW), pad_amount=3 (4 DW).
__device__ __forceinline__ void tdm_load_tile(const _Float16* gsrc, long rowStride,
                                              unsigned ldsOff) {
  unsigned long long ga = (unsigned long long)gsrc;
  v4u g0;
  g0.x = 1u;                                              // count=1, user mode
  g0.y = ldsOff;                                          // lds_addr (bytes)
  g0.z = (unsigned)(ga & 0xffffffffull);                  // global_addr[31:0]
  g0.w = (unsigned)((ga >> 32) & 0x1ffffffull) | (2u << 30); // addr[56:32] | type=2
  const unsigned td0 = 32u, td1 = 128u;
  unsigned long long s0 = (unsigned long long)rowStride;  // dim0 stride (elements)
  v8i g1;
  g1[0] = (int)((1u << 16) |            // data_size=1 (2 bytes)
                (1u << 20) |            // pad_enable
                (3u << 22) |            // pad_interval = 16 DWORDs (64B)
                (3u << 25));            // pad_amount   = 4 DWORDs (16B)
  g1[1] = (int)((td0 & 0xffffu) << 16);                       // tensor_dim0[15:0]
  g1[2] = (int)(((td0 >> 16) & 0xffffu) | ((td1 & 0xffffu) << 16)); // d0 hi | d1 lo
  g1[3] = (int)(((td1 >> 16) & 0xffffu) | (32u << 16));       // d1 hi | tile_dim0=32
  g1[4] = (int)(128u);                                        // tile_dim1=128, tile_dim2=0
  g1[5] = (int)(unsigned)(s0 & 0xffffffffull);                // dim0_stride[31:0]
  g1[6] = (int)(unsigned)((s0 >> 32) & 0xffffull);            // stride hi | d1_stride=0
  g1[7] = 0;
  v4i zz4 = {0, 0, 0, 0};
  v8i zz8 = {0, 0, 0, 0, 0, 0, 0, 0};
  // 6-arg toolchain form: (g0 v4u, g1 v8i, g2 v4i, g3 v4i, v8i, cpol)
  __builtin_amdgcn_tensor_load_to_lds(g0, g1, zz4, zz4, zz8, 0);
}
#endif

// ---------------------------------------------------------------------------
// WMMA GEMM:  C[b] (M x NC) = A[b] (M x K, row-major) * Bt[b]^T
// Bt stored (NC x K) row-major (pre-transposed, K-contiguous).
// 256 threads = 8 waves; block tile 128x128x32; wave tile 32x64 (2x4 WMMA).
// f16 inputs, f32 accumulate, optional f16 output. TDM-staged, double-buffered.
// ---------------------------------------------------------------------------
template<int M, int NC, int K, bool HALF_OUT>
__global__ __launch_bounds__(256) void gemm_wmma(
    const _Float16* __restrict__ A,  long strideA,
    const _Float16* __restrict__ Bt, long strideB,
    void* __restrict__ C,            long strideC)
{
  __shared__ __align__(16) _Float16 As[2][TILE_HALFS];
  __shared__ __align__(16) _Float16 Bs[2][TILE_HALFS];

  const int b   = blockIdx.y;
  const int nTN = NC / 128;
  const int bm  = (blockIdx.x / nTN) * 128;
  const int bn  = (blockIdx.x % nTN) * 128;
  const _Float16* Ab = A  + (long)b * strideA + (long)bm * K;
  const _Float16* Bb = Bt + (long)b * strideB + (long)bn * K;

  const int t    = threadIdx.x;
  const int lane = t & 31;
  const int wave = t >> 5;
  const int wm   = (wave & 3) * 32;   // 4 waves along M
  const int wn   = (wave >> 2) * 64;  // 2 waves along N
  const int r    = lane & 15;
  const int hi   = lane >> 4;
  constexpr int STEPS = K / 32;

  v8f acc[2][4] = {};

#ifdef USE_TDM
  const unsigned asOff = lds_off(&As[0][0]);
  const unsigned bsOff = lds_off(&Bs[0][0]);
  if (wave == 0) {                     // prologue: stage tile 0 into buffer 0
    tdm_load_tile(Ab, K, asOff);
    tdm_load_tile(Bb, K, bsOff);
  }
#endif

  for (int i = 0; i < STEPS; ++i) {
    const int cur = i & 1;
#ifdef USE_TDM
    if (wave == 0) {
      if (i + 1 < STEPS) {             // stage next tile into other buffer
        unsigned nb = (unsigned)((i + 1) & 1) * TILE_BYTES;
        tdm_load_tile(Ab + (long)(i + 1) * 32, K, asOff + nb);
        tdm_load_tile(Bb + (long)(i + 1) * 32, K, bsOff + nb);
        __builtin_amdgcn_s_wait_tensorcnt(2);   // pair i complete (in-order)
      } else {
        __builtin_amdgcn_s_wait_tensorcnt(0);
      }
    }
    __syncthreads();                   // publish buffer `cur` to all waves
#else
    // synchronous staging fallback (also used for host-side parse)
#pragma unroll
    for (int u = 0; u < 2; ++u) {
      int idx = t + u * 256;
      int row = idx >> 2;
      int ch  = (idx & 3) * 8;
      *(v8h*)&As[cur][row * LDS_STRIDE + ch] =
          *(const v8h*)&Ab[(long)row * K + i * 32 + ch];
      *(v8h*)&Bs[cur][row * LDS_STRIDE + ch] =
          *(const v8h*)&Bb[(long)row * K + i * 32 + ch];
    }
    if (i + 1 < STEPS) {
      __builtin_prefetch(&Ab[(long)(t >> 1) * K + i * 32 + 32], 0, 1);
      __builtin_prefetch(&Bb[(long)(t >> 1) * K + i * 32 + 32], 0, 1);
    }
    __syncthreads();
#endif

    // A fragments: 16-bit A 16x32 layout
    // lanes 0-15: M=lane, K 0-7 & 16-23 ; lanes 16-31: M=lane-16, K 8-15 & 24-31
    v16h af[2];
#pragma unroll
    for (int u = 0; u < 2; ++u) {
      const _Float16* p = &As[cur][(wm + u * 16 + r) * LDS_STRIDE];
      v8h lo = *(const v8h*)&p[hi * 8];
      v8h hb = *(const v8h*)&p[16 + hi * 8];
      af[u] = __builtin_shufflevector(lo, hb, 0,1,2,3,4,5,6,7,8,9,10,11,12,13,14,15);
    }
    // B fragments: 16-bit B 32x16 layout
    // lanes 0-15: N=lane, K 0-7 & 16-23 ; lanes 16-31: N=lane-16, K 8-15 & 24-31
    v16h bf[4];
#pragma unroll
    for (int j = 0; j < 4; ++j) {
      const _Float16* p = &Bs[cur][(wn + j * 16 + r) * LDS_STRIDE];
      v8h lo = *(const v8h*)&p[hi * 8];
      v8h hb = *(const v8h*)&p[16 + hi * 8];
      bf[j] = __builtin_shufflevector(lo, hb, 0,1,2,3,4,5,6,7,8,9,10,11,12,13,14,15);
    }
#pragma unroll
    for (int u = 0; u < 2; ++u)
#pragma unroll
      for (int j = 0; j < 4; ++j)
        acc[u][j] = __builtin_amdgcn_wmma_f32_16x16x32_f16(
            false, af[u], false, bf[j], (short)0, acc[u][j], false, false);
    __syncthreads();                   // compute done before buffer reuse
  }

  // epilogue: C 16x16 f32 layout: VGPR e -> M = e + 8*hi, N = lane&15
#pragma unroll
  for (int u = 0; u < 2; ++u)
#pragma unroll
    for (int j = 0; j < 4; ++j)
#pragma unroll
      for (int e = 0; e < 8; ++e) {
        int m = bm + wm + u * 16 + hi * 8 + e;
        int n = bn + wn + j * 16 + r;
        float v = acc[u][j][e];
        if (HALF_OUT)
          ((_Float16*)C)[(long)b * strideC + (long)m * NC + n] = (_Float16)v;
        else
          ((float*)C)[(long)b * strideC + (long)m * NC + n] = v;
      }
}

// ---------------------------------------------------------------------------
// Elementwise / reduction kernels
// ---------------------------------------------------------------------------
__global__ __launch_bounds__(256) void k_init(
    const float* __restrict__ p0, const float* __restrict__ q0,
    float* log_t, float* log_s, float* z,
    float* log_mu, float* log_eta, float* z1, float* z2)
{
  int i = blockIdx.x * 256 + threadIdx.x;
  int b = i >> 19;
  int rr = i & (N_ * D_ - 1);
  int n = rr >> 9;
  int d = rr & (D_ - 1);
  float p = p0[b * D_ + d];
  float q = q0[b * N_ + n];
  float lt = logf(q * p + EPSV);
  log_t[i] = lt; log_s[i] = lt; z[i] = 0.f;
  log_mu[i]  = logf(p);
  log_eta[i] = logf(q + EPSV);
  z1[i] = 0.f; z2[i] = 0.f;
}

__global__ __launch_bounds__(256) void k_cvt_c2(const float* __restrict__ c2,
                                                _Float16* __restrict__ c2h) {
  int i = blockIdx.x * 256 + threadIdx.x;
  c2h[i] = (_Float16)c2[i];
}

__global__ __launch_bounds__(256) void k_cvt_c1t(const float* __restrict__ c1,
                                                 _Float16* __restrict__ c1t) {
  int i = blockIdx.x * 256 + threadIdx.x;   // over B*D*D
  int b = i / (D_ * D_);
  int rr = i - b * D_ * D_;
  int row = rr >> 9;
  int col = rr & (D_ - 1);
  c1t[b * D_ * D_ + col * D_ + row] = (_Float16)c1[i];
}

// Et[b][d][n] = (f16) exp(src[b][n][d])  (transposed so GEMM "B" is K-major)
__global__ __launch_bounds__(256) void k_expT(const float* __restrict__ src,
                                              _Float16* __restrict__ Et) {
  int i = blockIdx.x * 256 + threadIdx.x;
  int b = i >> 19;
  int rr = i & (N_ * D_ - 1);
  int n = rr >> 9;
  int d = rr & (D_ - 1);
  Et[(long)b * D_ * N_ + (long)d * N_ + n] = (_Float16)__expf(src[i]);
}

// Row pass: y=(x+a0*tmp2-z)/rho+log_s ; log_t=log_mu-lse_d(y)+y ; sumt=sum_d exp(log_t)*mask
__global__ __launch_bounds__(256) void k_rowA(
    const float* __restrict__ x, const float* __restrict__ tmp2,
    const float* __restrict__ z, const float* __restrict__ log_s,
    const float* __restrict__ log_mu, const float* __restrict__ mask,
    float* __restrict__ log_t, float* __restrict__ sumt,
    const float* __restrict__ a0, const float* __restrict__ rho, int k)
{
  __shared__ float red[256];
  int n = blockIdx.x, b = blockIdx.y, t = threadIdx.x;
  long base = ((long)b * N_ + n) * D_;
  float A0 = a0[k], R = rho[k], invR = 1.f / R;
  int d0 = t, d1 = t + 256;
  float y0 = (x[base + d0] + A0 * tmp2[base + d0] - z[base + d0]) * invR + log_s[base + d0];
  float y1 = (x[base + d1] + A0 * tmp2[base + d1] - z[base + d1]) * invR + log_s[base + d1];
  red[t] = fmaxf(y0, y1); __syncthreads();
  for (int s = 128; s > 0; s >>= 1) { if (t < s) red[t] = fmaxf(red[t], red[t + s]); __syncthreads(); }
  float m = red[0]; __syncthreads();
  red[t] = __expf(y0 - m) + __expf(y1 - m); __syncthreads();
  for (int s = 128; s > 0; s >>= 1) { if (t < s) red[t] += red[t + s]; __syncthreads(); }
  float lse = m + __logf(red[0]); __syncthreads();
  float lt0 = log_mu[base + d0] - lse + y0;
  float lt1 = log_mu[base + d1] - lse + y1;
  log_t[base + d0] = lt0; log_t[base + d1] = lt1;
  float mk = mask[b * N_ + n];
  red[t] = (__expf(lt0) + __expf(lt1)) * mk; __syncthreads();
  for (int s = 128; s > 0; s >>= 1) { if (t < s) red[t] += red[t + s]; __syncthreads(); }
  if (t == 0) sumt[b * N_ + n] = red[0];
}

// Column pass: y=(z+a0*tmp2+rho*log_t)/(a1+rho) ; log_s=log_eta-lse_n(y)+y ;
// z += rho*(exp(log_t)-exp(log_s))*mask ; sums[b,d]=sum_n exp(log_s)*mask
__global__ __launch_bounds__(256) void k_colS(
    const float* __restrict__ tmp2, float* __restrict__ z,
    const float* __restrict__ log_t, const float* __restrict__ log_eta,
    const float* __restrict__ mask, float* __restrict__ log_s,
    float* __restrict__ sums,
    const float* __restrict__ a0, const float* __restrict__ a1,
    const float* __restrict__ rho, int k)
{
  __shared__ float Ms[256], Ss[256];
  int t = threadIdx.x;
  int dl = t & 63, q = t >> 6;
  int b = blockIdx.y;
  int d = blockIdx.x * 64 + dl;
  float A0 = a0[k], A1 = a1[k], R = rho[k];
  float inv = 1.f / (A1 + R);
  long colBase = (long)b * N_ * D_ + d;
  float m = -3.4e38f, s = 0.f;
  for (int n = q * 256; n < q * 256 + 256; ++n) {
    long idx = colBase + (long)n * D_;
    float y = (z[idx] + A0 * tmp2[idx] + R * log_t[idx]) * inv;
    if (y > m) { s = s * __expf(m - y) + 1.f; m = y; } else { s += __expf(y - m); }
  }
  Ms[t] = m; Ss[t] = s; __syncthreads();
  float M2 = fmaxf(fmaxf(Ms[dl], Ms[dl + 64]), fmaxf(Ms[dl + 128], Ms[dl + 192]));
  float S2 = Ss[dl] * __expf(Ms[dl] - M2) + Ss[dl + 64] * __expf(Ms[dl + 64] - M2)
           + Ss[dl + 128] * __expf(Ms[dl + 128] - M2) + Ss[dl + 192] * __expf(Ms[dl + 192] - M2);
  float lse = M2 + __logf(S2);
  __syncthreads();
  float mysum = 0.f;
  for (int n = q * 256; n < q * 256 + 256; ++n) {
    long idx = colBase + (long)n * D_;
    float y  = (z[idx] + A0 * tmp2[idx] + R * log_t[idx]) * inv;
    float ls = log_eta[idx] - lse + y;
    log_s[idx] = ls;
    float mk = mask[b * N_ + n];
    float sv = __expf(ls) * mk;
    float tv = __expf(log_t[idx]) * mk;
    z[idx] = z[idx] + R * (tv - sv);
    mysum += sv;
  }
  Ss[t] = mysum; __syncthreads();
  if (q == 0) sums[b * D_ + d] = Ss[dl] + Ss[dl + 64] + Ss[dl + 128] + Ss[dl + 192];
}

// mu update (row-LSE over D): y=(R*log_mu+A2*log_p0-z1)/(R+A2); log_mu=y-lse;
// z1 += R*(exp(log_mu)-sumt[b,n])
__global__ __launch_bounds__(256) void k_mu(
    const float* __restrict__ p0, float* __restrict__ log_mu,
    float* __restrict__ z1, const float* __restrict__ sumt,
    const float* __restrict__ a2, const float* __restrict__ rho, int k)
{
  __shared__ float red[256];
  int n = blockIdx.x, b = blockIdx.y, t = threadIdx.x;
  long base = ((long)b * N_ + n) * D_;
  float A2 = a2[k], R = rho[k];
  float inv = 1.f / (R + A2);
  int d0 = t, d1 = t + 256;
  float y0 = (R * log_mu[base + d0] + A2 * __logf(p0[b * D_ + d0]) - z1[base + d0]) * inv;
  float y1 = (R * log_mu[base + d1] + A2 * __logf(p0[b * D_ + d1]) - z1[base + d1]) * inv;
  red[t] = fmaxf(y0, y1); __syncthreads();
  for (int s = 128; s > 0; s >>= 1) { if (t < s) red[t] = fmaxf(red[t], red[t + s]); __syncthreads(); }
  float m = red[0]; __syncthreads();
  red[t] = __expf(y0 - m) + __expf(y1 - m); __syncthreads();
  for (int s = 128; s > 0; s >>= 1) { if (t < s) red[t] += red[t + s]; __syncthreads(); }
  float lse = m + __logf(red[0]);
  float st = sumt[b * N_ + n];
  float nm0 = y0 - lse, nm1 = y1 - lse;
  log_mu[base + d0] = nm0; log_mu[base + d1] = nm1;
  z1[base + d0] += R * (__expf(nm0) - st);
  z1[base + d1] += R * (__expf(nm1) - st);
}

// eta update (column-LSE over N): y=(R*log_eta+A3*log_q0-z2)/(R+A3); log_eta=y-lse;
// z2 += R*(exp(log_eta)-sums[b,d])
__global__ __launch_bounds__(256) void k_eta(
    const float* __restrict__ q0, float* __restrict__ log_eta,
    float* __restrict__ z2, const float* __restrict__ sums,
    const float* __restrict__ a3, const float* __restrict__ rho, int k)
{
  __shared__ float Ms[256], Ss[256];
  int t = threadIdx.x;
  int dl = t & 63, q = t >> 6;
  int b = blockIdx.y;
  int d = blockIdx.x * 64 + dl;
  float A3 = a3[k], R = rho[k];
  float inv = 1.f / (R + A3);
  long colBase = (long)b * N_ * D_ + d;
  float m = -3.4e38f, s = 0.f;
  for (int n = q * 256; n < q * 256 + 256; ++n) {
    long idx = colBase + (long)n * D_;
    float lq = __logf(q0[b * N_ + n] + EPSV);
    float y = (R * log_eta[idx] + A3 * lq - z2[idx]) * inv;
    if (y > m) { s = s * __expf(m - y) + 1.f; m = y; } else { s += __expf(y - m); }
  }
  Ms[t] = m; Ss[t] = s; __syncthreads();
  float M2 = fmaxf(fmaxf(Ms[dl], Ms[dl + 64]), fmaxf(Ms[dl + 128], Ms[dl + 192]));
  float S2 = Ss[dl] * __expf(Ms[dl] - M2) + Ss[dl + 64] * __expf(Ms[dl + 64] - M2)
           + Ss[dl + 128] * __expf(Ms[dl + 128] - M2) + Ss[dl + 192] * __expf(Ms[dl + 192] - M2);
  float lse = M2 + __logf(S2);
  __syncthreads();
  float sm = sums[b * D_ + d];
  for (int n = q * 256; n < q * 256 + 256; ++n) {
    long idx = colBase + (long)n * D_;
    float lq = __logf(q0[b * N_ + n] + EPSV);
    float y  = (R * log_eta[idx] + A3 * lq - z2[idx]) * inv;
    float ne = y - lse;
    log_eta[idx] = ne;
    z2[idx] = z2[idx] + R * (__expf(ne) - sm);
  }
}

__global__ __launch_bounds__(256) void k_final(const float* __restrict__ log_t,
                                               const float* __restrict__ mask,
                                               float* __restrict__ out) {
  int i = blockIdx.x * 256 + threadIdx.x;
  int b = i >> 19;
  int n = (i & (N_ * D_ - 1)) >> 9;
  out[i] = __expf(log_t[i]) * mask[b * N_ + n];
}

// ---------------------------------------------------------------------------
extern "C" void kernel_launch(void* const* d_in, const int* in_sizes, int n_in,
                              void* d_out, int out_size, void* d_ws, size_t ws_size,
                              hipStream_t stream) {
  const float* x    = (const float*)d_in[0];
  const float* c1   = (const float*)d_in[1];
  const float* c2   = (const float*)d_in[2];
  const float* p0   = (const float*)d_in[3];
  const float* q0   = (const float*)d_in[4];
  const float* a0   = (const float*)d_in[5];
  const float* a1   = (const float*)d_in[6];
  const float* a2   = (const float*)d_in[7];
  const float* a3   = (const float*)d_in[8];
  const float* rho  = (const float*)d_in[9];
  const float* mask = (const float*)d_in[10];
  // d_in[11] = num (device scalar); reference fixes num=4.

  char* w = (char*)d_ws;
  const size_t F = (size_t)BND * 4;   // 16 MB
  float*    log_t   = (float*)(w);
  float*    log_s   = (float*)(w + 1 * F);
  float*    z       = (float*)(w + 2 * F);
  float*    log_mu  = (float*)(w + 3 * F);
  float*    log_eta = (float*)(w + 4 * F);
  float*    z1      = (float*)(w + 5 * F);
  float*    z2      = (float*)(w + 6 * F);
  float*    tmp2    = (float*)(w + 7 * F);
  _Float16* T1h     = (_Float16*)(w + 8 * F);                    // 8 MB
  _Float16* Eth     = (_Float16*)(w + 8 * F + F / 2);            // 8 MB
  _Float16* c2h     = (_Float16*)(w + 9 * F);                    // 16 MB
  _Float16* c1t     = (_Float16*)(w + 10 * F);                   // 4 MB
  float*    sumt    = (float*)(w + 10 * F + F / 4);
  float*    sums    = (float*)(w + 10 * F + F / 4 + B_ * N_ * 4);

  dim3 blk(256);
  k_init<<<BND / 256, blk, 0, stream>>>(p0, q0, log_t, log_s, z, log_mu, log_eta, z1, z2);
  k_cvt_c2<<<(B_ * N_ * N_) / 256, blk, 0, stream>>>(c2, c2h);
  k_cvt_c1t<<<(B_ * D_ * D_) / 256, blk, 0, stream>>>(c1, c1t);

  dim3 gGemm(32, B_);   // (1024/128)*(512/128) = 32 tiles per batch
  for (int k = 0; k < 4; ++k) {
    // --- log_t update ---
    k_expT<<<BND / 256, blk, 0, stream>>>(log_s, Eth);
    gemm_wmma<N_, D_, N_, true ><<<gGemm, blk, 0, stream>>>(
        c2h, (long)N_ * N_, Eth, (long)D_ * N_, T1h, (long)N_ * D_);
    gemm_wmma<N_, D_, D_, false><<<gGemm, blk, 0, stream>>>(
        T1h, (long)N_ * D_, c1t, (long)D_ * D_, tmp2, (long)N_ * D_);
    k_rowA<<<dim3(N_, B_), blk, 0, stream>>>(x, tmp2, z, log_s, log_mu, mask,
                                             log_t, sumt, a0, rho, k);
    // --- log_s update ---
    k_expT<<<BND / 256, blk, 0, stream>>>(log_t, Eth);
    gemm_wmma<N_, D_, N_, true ><<<gGemm, blk, 0, stream>>>(
        c2h, (long)N_ * N_, Eth, (long)D_ * N_, T1h, (long)N_ * D_);
    gemm_wmma<N_, D_, D_, false><<<gGemm, blk, 0, stream>>>(
        T1h, (long)N_ * D_, c1t, (long)D_ * D_, tmp2, (long)N_ * D_);
    k_colS<<<dim3(D_ / 64, B_), blk, 0, stream>>>(tmp2, z, log_t, log_eta, mask,
                                                  log_s, sums, a0, a1, rho, k);
    // --- marginal updates ---
    k_mu<<<dim3(N_, B_), blk, 0, stream>>>(p0, log_mu, z1, sumt, a2, rho, k);
    k_eta<<<dim3(D_ / 64, B_), blk, 0, stream>>>(q0, log_eta, z2, sums, a3, rho, k);
  }
  k_final<<<BND / 256, blk, 0, stream>>>(log_t, mask, (float*)d_out);
}
